// Fp8LightingIndexerDecodeLayer_16965120819287
// MI455X (gfx1250) — 8-bit matrix operand order repaired and checked by reading, compiled, not run on hardware
//
#include <hip/hip_runtime.h>
#include <hip/hip_bf16.h>
#include <math.h>

typedef __attribute__((ext_vector_type(16))) int   v16i;
typedef __attribute__((ext_vector_type(8)))  float v8f;
typedef __attribute__((ext_vector_type(4)))  float v4f;

#define SQ      64
#define HQ      64
#define DD      128
#define TT      32768
#define QROWS   (SQ * HQ)
#define FP8MAXF 448.0f

__device__ __forceinline__ unsigned int f32_to_e4m3_sw(float x) {
    unsigned int u    = __float_as_uint(x);
    unsigned int sign = (u >> 24) & 0x80u;
    float ax = fabsf(x);
    if (ax != ax) return sign | 0x7Fu;
    if (ax == 0.0f) return sign;
    if (ax >= FP8MAXF) return sign | 0x7Eu;
    unsigned int v   = __float_as_uint(ax);
    unsigned int lsb = (v >> 20) & 1u;
    v += 0x0007FFFFu + lsb;
    int e          = (int)(v >> 23) - 127;
    unsigned int m = (v >> 20) & 7u;
    unsigned int r;
    if (e < -6) {
        unsigned int mi = (unsigned int)rintf(ax * 512.0f);
        r = (mi > 7u) ? 8u : mi;
    } else if (e > 8 || (e == 8 && m > 6u)) {
        r = 0x7Eu;
    } else {
        r = ((unsigned int)(e + 7) << 3) | m;
    }
    return sign | r;
}

__device__ __forceinline__ unsigned int f32x4_to_e4m3x4(float x0, float x1,
                                                        float x2, float x3) {
#if __has_builtin(__builtin_amdgcn_cvt_pk_fp8_f32)
    int p = __builtin_amdgcn_cvt_pk_fp8_f32(x0, x1, 0, false);
    p     = __builtin_amdgcn_cvt_pk_fp8_f32(x2, x3, p, true);
    return (unsigned int)p;
#else
    return  f32_to_e4m3_sw(x0)        | (f32_to_e4m3_sw(x1) << 8) |
           (f32_to_e4m3_sw(x2) << 16) | (f32_to_e4m3_sw(x3) << 24);
#endif
}

__global__ void __launch_bounds__(256)
fp8_quant_kernel(const float* __restrict__ index_q,
                 const float* __restrict__ index_k,
                 const float* __restrict__ weights,
                 unsigned char* __restrict__ k8,
                 unsigned char* __restrict__ q8,
                 float* __restrict__ kscale,
                 float* __restrict__ qsw) {
    const int wavesPerBlock = blockDim.x >> 5;
    const int wid  = blockIdx.x * wavesPerBlock + (threadIdx.x >> 5);
    const int lane = threadIdx.x & 31;
    const int totalRows = TT + QROWS;
    if (wid >= totalRows) return;

    const bool isK = (wid < TT);
    const int  qr  = wid - TT;
    const float* src = isK ? (index_k + (size_t)wid * DD)
                           : (index_q + (size_t)qr  * DD);

    v4f x = *(const v4f*)(src + lane * 4);
    float amax = fmaxf(fmaxf(fabsf(x.x), fabsf(x.y)),
                       fmaxf(fabsf(x.z), fabsf(x.w)));
#pragma unroll
    for (int off = 16; off >= 1; off >>= 1)
        amax = fmaxf(amax, __shfl_xor(amax, off, 32));

    const float scale = fmaxf(amax * (1.0f / FP8MAXF), 1e-12f);
    const float inv   = 1.0f / scale;

    unsigned int packed = f32x4_to_e4m3x4(x.x * inv, x.y * inv,
                                          x.z * inv, x.w * inv);
    unsigned char* dst = isK ? (k8 + (size_t)wid * DD)
                             : (q8 + (size_t)qr  * DD);
    ((unsigned int*)dst)[lane] = packed;

    if (lane == 0) {
        if (isK) kscale[wid] = scale;
        else     qsw[qr]     = 0.5f * scale * weights[qr];
    }
}

__global__ void __launch_bounds__(256)
fp8_index_score_kernel(const unsigned char* __restrict__ q8,
                       const unsigned char* __restrict__ k8,
                       const float* __restrict__ qsw,
                       const float* __restrict__ kscale,
                       const int*   __restrict__ cu_ks,
                       const int*   __restrict__ cu_ke,
                       float*       __restrict__ out) {
    const int wave  = threadIdx.x >> 5;
    const int lane  = threadIdx.x & 31;
    const int chunk = blockIdx.x & 255;
    const int s     = ((blockIdx.x >> 8) << 3) + wave;
    const int t0base = chunk * 128;

    const int hi    = (lane >= 16) ? 8 : 0;
    const int nlane = lane & 15;

    v16i a[4];
#pragma unroll
    for (int ht = 0; ht < 4; ++ht) {
        const unsigned char* qrow =
            q8 + (size_t)(s * HQ + ht * 16 + nlane) * DD;
#pragma unroll
        for (int v = 0; v < 16; ++v) {
            const int K = 32 * (v >> 2) + 2 * hi + 4 * (v & 3);
            a[ht][v] = *(const int*)(qrow + K);
        }
    }

    float qw[4][8];
#pragma unroll
    for (int ht = 0; ht < 4; ++ht)
#pragma unroll
        for (int r = 0; r < 8; ++r)
            qw[ht][r] = qsw[s * HQ + ht * 16 + hi + r];

    const int ksS = cu_ks[s];
    const int keS = cu_ke[s];

#pragma unroll 1
    for (int tt = 0; tt < 8; ++tt) {
        const int t0 = t0base + tt * 16;

        const unsigned char* krow = k8 + (size_t)(t0 + nlane) * DD;
        v16i b;
#pragma unroll
        for (int v = 0; v < 16; ++v) {
            const int K = 32 * (v >> 2) + ((lane >= 16) ? 16 : 0) + 4 * (v & 3);
            b[v] = *(const int*)(krow + K);
        }
        __builtin_prefetch(krow + 16 * DD, 0, 1);

        float acc[4] = {0.f, 0.f, 0.f, 0.f};
#pragma unroll
        for (int ht = 0; ht < 4; ++ht) {
            v8f c = {};
            c = __builtin_amdgcn_wmma_f32_16x16x128_fp8_fp8(
                    a[ht], b, (short)0, c, false, false);
#pragma unroll
            for (int r = 0; r < 8; ++r) {
                const float t = c[r] + fabsf(c[r]);
                acc[r & 3] += t * qw[ht][r];
            }
        }
        float part = (acc[0] + acc[1]) + (acc[2] + acc[3]);
        part *= kscale[t0 + nlane];
        part += __shfl_xor(part, 16, 32);

        if (lane < 16) {
            const int  t   = t0 + lane;
            const bool vis = (t >= ksS) && (t < keS);
            out[(size_t)s * TT + t] = vis ? part : -INFINITY;
        }
    }
}

extern "C" void kernel_launch(void* const* d_in, const int* in_sizes, int n_in,
                              void* d_out, int out_size, void* d_ws, size_t ws_size,
                              hipStream_t stream) {
    const float* index_q = (const float*)d_in[0];
    const float* index_k = (const float*)d_in[1];
    const float* weights = (const float*)d_in[2];
    const int*   cu_ks   = (const int*)d_in[3];
    const int*   cu_ke   = (const int*)d_in[4];
    float*       out     = (float*)d_out;

    unsigned char* ws = (unsigned char*)d_ws;
    unsigned char* k8 = ws;
    unsigned char* q8 = ws + (size_t)TT * DD;
    float* kscale = (float*)(ws + (size_t)TT * DD + (size_t)QROWS * DD);
    float* qsw    = kscale + TT;

    {
        const int rows   = TT + QROWS;
        const int blocks = (rows + 7) / 8;
        fp8_quant_kernel<<<blocks, 256, 0, stream>>>(
            index_q, index_k, weights, k8, q8, kscale, qsw);
    }
    {
        fp8_index_score_kernel<<<8 * 256, 256, 0, stream>>>(
            q8, k8, qsw, kscale, cu_ks, cu_ke, out);
    }
}
